// MultiHeadAttention_53463752901078
// MI455X (gfx1250) — matrix operand order repaired and checked by reading, not compiled, not run on hardware
//
#include <hip/hip_runtime.h>
#include <cstdint>

// ---------------------------------------------------------------------------
// MI455X / gfx1250 multi-head attention, bf16 WMMA (f32 accumulate) pipeline.
// Async global->LDS double buffering for K/V tiles + ds_swizzle softmax
// reductions.
// ---------------------------------------------------------------------------

typedef __bf16 bf16_t;
typedef __attribute__((ext_vector_type(4)))  bf16_t bf16x4;
typedef __attribute__((ext_vector_type(8)))  bf16_t bf16x8;
typedef __attribute__((ext_vector_type(16))) bf16_t v16bf;
typedef __attribute__((ext_vector_type(8)))  float  v8f;
typedef int v4i __attribute__((vector_size(16)));   // matches builtin prototype

#define GLOBAL_AS __attribute__((address_space(1)))
#define LDS_AS    __attribute__((address_space(3)))

#if defined(__gfx1250__) && __has_builtin(__builtin_amdgcn_global_load_async_to_lds_b128)
#define USE_ASYNC_LDS 1
#else
#define USE_ASYNC_LDS 0
#endif

constexpr int BATCH = 4;
constexpr int SEQ   = 2048;
constexpr int DM    = 1024;
constexpr int HEADS = 16;
constexpr int DK    = 64;
constexpr int MROWS = BATCH * SEQ;   // 8192

union FragU { v16bf v; bf16x8 h[2]; };

// A-matrix fragment (16x32 bf16, MxK). Lanes 0-15: K0-7 & K16-23; lanes 16-31:
// K8-15 & K24-31 (ISA 7.12.2 "16-bit A-Matrix 16x32").
__device__ __forceinline__ v16bf load_frag_a(const bf16_t* rowp, int kbase, int hi) {
  FragU f;
  f.h[0] = *(const bf16x8*)(rowp + kbase + hi * 8);
  f.h[1] = *(const bf16x8*)(rowp + kbase + 16 + hi * 8);
  return f.v;
}

// B-matrix fragment (32x16 bf16, KxN). Lane holds N=lane&15; lanes 0-15 carry
// K0-7 & K16-23, lanes 16-31 carry K8-15 & K24-31 (same K order as A).
__device__ __forceinline__ v16bf load_frag_b(const bf16_t* rowp, int kbase, int hi) {
  FragU f;
  const bf16_t* p = rowp + kbase + hi * 8;
  f.h[0] = *(const bf16x8*)(p);
  f.h[1] = *(const bf16x8*)(p + 16);
  return f.v;
}

__device__ __forceinline__ v8f wmma_bf16(v16bf a, v16bf b, v8f c) {
  return __builtin_amdgcn_wmma_f32_16x16x32_bf16(
      /*neg_a=*/false, a, /*neg_b=*/false, b,
      /*c_mod=*/(short)0, c, /*reuse_a=*/false, /*reuse_b=*/false);
}

__device__ __forceinline__ v8f zero8() {
  v8f z = {0.f, 0.f, 0.f, 0.f, 0.f, 0.f, 0.f, 0.f};
  return z;
}

// XOR butterfly across lanes via ds_swizzle (group-of-32 mode: and=0x1f, xor=MASK)
template <int MASK>
__device__ __forceinline__ float swz_xor(float x) {
  int i = __builtin_amdgcn_ds_swizzle(__builtin_bit_cast(int, x), 0x1f | (MASK << 10));
  return __builtin_bit_cast(float, i);
}

#if USE_ASYNC_LDS
template <int N>
__device__ __forceinline__ void async_wait() {
#if __has_builtin(__builtin_amdgcn_s_wait_asynccnt)
  __builtin_amdgcn_s_wait_asynccnt((unsigned short)N);
#else
  asm volatile("s_wait_asynccnt %0" ::"n"(N) : "memory");
#endif
}
// 16B async copy global -> LDS (tracked by ASYNCcnt, in-order completion)
__device__ __forceinline__ void async_cp16(const bf16_t* g, bf16_t* l) {
  __builtin_amdgcn_global_load_async_to_lds_b128(
      (GLOBAL_AS v4i*)(uintptr_t)g, (LDS_AS v4i*)(uintptr_t)l, 0, 0);
}
#endif

enum { MODE_QK = 0, MODE_V = 1, MODE_OUT = 2 };

// ---------------------------------------------------------------------------
// Y = X @ W^T + b. 128x128 block tile, 8 waves, each wave 64x32 (4x2 wmma
// tiles), K-step 32. fp32 operands are converted to bf16 while staging into
// LDS (so no async copy here -- conversion needs the VALU anyway).
// ---------------------------------------------------------------------------
template <bool A_BF16>
__global__ __launch_bounds__(256)
void gemm_proj(const float* __restrict__ Af, const bf16_t* __restrict__ Ab,
               const float* __restrict__ W, const float* __restrict__ bias,
               bf16_t* __restrict__ ws_out, float* __restrict__ f_out,
               int mode, float scale) {
  constexpr int LD = 40;                       // padded bf16 row stride (80B, 16B aligned)
  __shared__ bf16_t Alds[128 * LD];
  __shared__ bf16_t Blds[128 * LD];

  const int tid  = threadIdx.x;
  const int lane = tid & 31;
  const int w    = tid >> 5;
  const int hi   = lane >> 4;
  const int ln   = lane & 15;
  const int mblk = blockIdx.y * 128;
  const int nblk = blockIdx.x * 128;
  const int mbase = (w & 1) * 64;
  const int nbase = (w >> 1) * 32;

  v8f acc[4][2];
  for (int mt = 0; mt < 4; ++mt)
    for (int nt = 0; nt < 2; ++nt) acc[mt][nt] = zero8();

  for (int k0 = 0; k0 < DM; k0 += 32) {
    if (A_BF16) {
      for (int i = 0; i < 2; ++i) {
        int c = tid + i * 256;
        int row = c >> 2, kc = (c & 3) * 8;
        *(bf16x8*)&Alds[row * LD + kc] =
            *(const bf16x8*)(Ab + (size_t)(mblk + row) * DM + k0 + kc);
      }
    } else {
      for (int i = 0; i < 4; ++i) {
        int c = tid + i * 256;
        int row = c >> 3, kc = (c & 7) * 4;
        const float4 a = *(const float4*)(Af + (size_t)(mblk + row) * DM + k0 + kc);
        bf16x4 hq = {(bf16_t)a.x, (bf16_t)a.y, (bf16_t)a.z, (bf16_t)a.w};
        *(bf16x4*)&Alds[row * LD + kc] = hq;
      }
    }
    for (int i = 0; i < 4; ++i) {
      int c = tid + i * 256;
      int row = c >> 3, kc = (c & 7) * 4;
      const float4 a = *(const float4*)(W + (size_t)(nblk + row) * DM + k0 + kc);
      bf16x4 hq = {(bf16_t)a.x, (bf16_t)a.y, (bf16_t)a.z, (bf16_t)a.w};
      *(bf16x4*)&Blds[row * LD + kc] = hq;
    }
    __syncthreads();

    v16bf bfrag[2];
    for (int nt = 0; nt < 2; ++nt)
      bfrag[nt] = load_frag_b(&Blds[(nbase + nt * 16 + ln) * LD], 0, hi);
    for (int mt = 0; mt < 4; ++mt) {
      v16bf afrag = load_frag_a(&Alds[(mbase + mt * 16 + ln) * LD], 0, hi);
      for (int nt = 0; nt < 2; ++nt)
        acc[mt][nt] = wmma_bf16(afrag, bfrag[nt], acc[mt][nt]);
    }
    __syncthreads();
  }

  for (int mt = 0; mt < 4; ++mt)
    for (int nt = 0; nt < 2; ++nt) {
      int gn = nblk + nbase + nt * 16 + ln;
      float bv = bias[gn];
      for (int vv = 0; vv < 8; ++vv) {
        int gm = mblk + mbase + mt * 16 + vv + hi * 8;
        float y = (acc[mt][nt][vv] + bv) * scale;
        if (mode == MODE_OUT) {
          f_out[(size_t)gm * DM + gn] = y;
        } else {
          int bb = gm >> 11, ss = gm & (SEQ - 1);
          int hh = gn >> 6,  dd = gn & (DK - 1);
          if (mode == MODE_QK)   // [B,H,S,64]
            ws_out[(((size_t)(bb * HEADS + hh)) * SEQ + ss) * DK + dd] = (bf16_t)y;
          else                   // V transposed: [B,H,64,S]
            ws_out[(((size_t)(bb * HEADS + hh)) * DK + dd) * SEQ + ss] = (bf16_t)y;
        }
      }
    }
}

// ---------------------------------------------------------------------------
// Flash attention: grid (S/64 q-tiles, B*H). 128 threads = 4 waves; wave w
// owns 16 query rows. K/V tiles double-buffered; staged with
// GLOBAL_LOAD_ASYNC_TO_LDS_B128 when available (ASYNCcnt pipelined one tile
// ahead), else synchronous b128 copies.
// ---------------------------------------------------------------------------
__global__ __launch_bounds__(128)
void flash_attn(const bf16_t* __restrict__ Q, const bf16_t* __restrict__ K,
                const bf16_t* __restrict__ V, bf16_t* __restrict__ O) {
  __shared__ bf16_t Qt[64 * DK];          // [q][d]
  __shared__ bf16_t Kt[2][64 * DK];       // [key][d]
  __shared__ bf16_t Vt[2][DK * 64];       // [d][key]
  __shared__ bf16_t Pt[4][16 * 64];       // per-wave P tile

  const int tid  = threadIdx.x;
  const int lane = tid & 31;
  const int w    = tid >> 5;
  const int hi   = lane >> 4;
  const int ln   = lane & 15;

  const size_t head = blockIdx.y;                       // b*H + h
  const bf16_t* qh = Q + head * (size_t)SEQ * DK;
  const bf16_t* kh = K + head * (size_t)SEQ * DK;
  const bf16_t* vh = V + head * (size_t)DK * SEQ;       // [d][s]
  const int q0 = blockIdx.x * 64;
  constexpr int NT = SEQ / 64;

  // stage K/V tile kt into LDS buffer `buf` (8 x b128 per thread)
  auto stage = [&](int kt, int buf) {
    const int k0 = kt * 64;
    for (int i = 0; i < 4; ++i) {
      int c = tid + i * 128;
      int row = c >> 3, kc = (c & 7) * 8;
#if USE_ASYNC_LDS
      async_cp16(kh + (size_t)(k0 + row) * DK + kc, &Kt[buf][row * DK + kc]);
      async_cp16(vh + (size_t)row * SEQ + k0 + kc,  &Vt[buf][row * DK + kc]);
#else
      *(bf16x8*)&Kt[buf][row * DK + kc] =
          *(const bf16x8*)(kh + (size_t)(k0 + row) * DK + kc);
      *(bf16x8*)&Vt[buf][row * DK + kc] =
          *(const bf16x8*)(vh + (size_t)row * SEQ + k0 + kc);
#endif
    }
  };

  stage(0, 0);   // prologue: tile 0 in flight

  for (int i = 0; i < 4; ++i) {
    int c = tid + i * 128;
    int row = c >> 3, kc = (c & 7) * 8;
    *(bf16x8*)&Qt[row * DK + kc] = *(const bf16x8*)(qh + (size_t)(q0 + row) * DK + kc);
  }
  __syncthreads();

  v16bf qf[2];
  for (int kd = 0; kd < 2; ++kd)
    qf[kd] = load_frag_a(&Qt[(w * 16 + ln) * DK], kd * 32, hi);

  float mrow[8], lrow[8];
  v8f oacc[4];
  for (int i = 0; i < 8; ++i) { mrow[i] = -1e30f; lrow[i] = 0.f; }
  for (int dt = 0; dt < 4; ++dt) oacc[dt] = zero8();

  for (int kt = 0; kt < NT; ++kt) {
    const int cur = kt & 1;
    if (kt + 1 < NT) stage(kt + 1, cur ^ 1);    // prefetch next tile
#if USE_ASYNC_LDS
    if (kt + 1 < NT) async_wait<8>();           // in-order: tile kt resident
    else             async_wait<0>();
#else
    if (kt + 1 < NT) {
      __builtin_prefetch(kh + (size_t)(kt + 1) * 64 * DK + tid * 32, 0, 1);
    }
#endif
    __syncthreads();                            // all threads' tile data visible

    // scores: S = Q @ K^T (contract over d)
    v8f sc[4];
    for (int nt = 0; nt < 4; ++nt) {
      sc[nt] = zero8();
      for (int kd = 0; kd < 2; ++kd) {
        v16bf bf = load_frag_b(&Kt[cur][(nt * 16 + ln) * DK], kd * 32, hi);
        sc[nt] = wmma_bf16(qf[kd], bf, sc[nt]);
      }
    }

    // online softmax over this 64-key block (rows live on 16-lane half-waves)
    float mn[8];
    for (int vv = 0; vv < 8; ++vv)
      mn[vv] = fmaxf(fmaxf(sc[0][vv], sc[1][vv]), fmaxf(sc[2][vv], sc[3][vv]));
    for (int vv = 0; vv < 8; ++vv) mn[vv] = fmaxf(mn[vv], swz_xor<1>(mn[vv]));
    for (int vv = 0; vv < 8; ++vv) mn[vv] = fmaxf(mn[vv], swz_xor<2>(mn[vv]));
    for (int vv = 0; vv < 8; ++vv) mn[vv] = fmaxf(mn[vv], swz_xor<4>(mn[vv]));
    for (int vv = 0; vv < 8; ++vv) mn[vv] = fmaxf(mn[vv], swz_xor<8>(mn[vv]));

    float alpha[8], rs[8];
    for (int vv = 0; vv < 8; ++vv) {
      mn[vv]    = fmaxf(mn[vv], mrow[vv]);
      alpha[vv] = __expf(mrow[vv] - mn[vv]);
      mrow[vv]  = mn[vv];
      rs[vv]    = 0.f;
    }
    for (int nt = 0; nt < 4; ++nt)
      for (int vv = 0; vv < 8; ++vv) {
        float p = __expf(sc[nt][vv] - mn[vv]);
        sc[nt][vv] = p;
        rs[vv] += p;
      }
    for (int vv = 0; vv < 8; ++vv) rs[vv] += swz_xor<1>(rs[vv]);
    for (int vv = 0; vv < 8; ++vv) rs[vv] += swz_xor<2>(rs[vv]);
    for (int vv = 0; vv < 8; ++vv) rs[vv] += swz_xor<4>(rs[vv]);
    for (int vv = 0; vv < 8; ++vv) rs[vv] += swz_xor<8>(rs[vv]);
    for (int vv = 0; vv < 8; ++vv) lrow[vv] = lrow[vv] * alpha[vv] + rs[vv];
    for (int dt = 0; dt < 4; ++dt)
      for (int vv = 0; vv < 8; ++vv) oacc[dt][vv] *= alpha[vv];

    // spill P (C layout) to LDS so it can be re-read in A layout
    for (int nt = 0; nt < 4; ++nt)
      for (int vv = 0; vv < 8; ++vv)
        Pt[w][(vv + hi * 8) * 64 + nt * 16 + ln] = (bf16_t)sc[nt][vv];
    __syncthreads();

    // O += P @ V (contract over keys; Vt rows are d, K-contiguous keys)
    for (int kc = 0; kc < 2; ++kc) {
      v16bf af = load_frag_a(&Pt[w][ln * 64], kc * 32, hi);
      for (int dt = 0; dt < 4; ++dt) {
        v16bf bf = load_frag_b(&Vt[cur][(dt * 16 + ln) * 64], kc * 32, hi);
        oacc[dt] = wmma_bf16(af, bf, oacc[dt]);
      }
    }
    __syncthreads();        // reads of Kt/Vt[cur] & Pt done before reuse
  }

  // epilogue: O/l, write bf16 as [B,S,H*dk] ready for the output projection
  const int bb = (int)head >> 4;
  const int hh = (int)head & 15;
  for (int dt = 0; dt < 4; ++dt)
    for (int vv = 0; vv < 8; ++vv) {
      int row = q0 + w * 16 + vv + hi * 8;
      int dd  = dt * 16 + ln;
      float y = oacc[dt][vv] / lrow[vv];
      O[((size_t)(bb * SEQ + row)) * DM + hh * DK + dd] = (bf16_t)y;
    }
}

// ---------------------------------------------------------------------------
extern "C" void kernel_launch(void* const* d_in, const int* in_sizes, int n_in,
                              void* d_out, int out_size, void* d_ws, size_t ws_size,
                              hipStream_t stream) {
  (void)in_sizes; (void)n_in; (void)out_size; (void)ws_size;
  const float* query = (const float*)d_in[0];
  const float* key_  = (const float*)d_in[1];
  const float* value = (const float*)d_in[2];
  const float* Wq = (const float*)d_in[3];
  const float* bq = (const float*)d_in[4];
  const float* Wk = (const float*)d_in[5];
  const float* bk = (const float*)d_in[6];
  const float* Wv = (const float*)d_in[7];
  const float* bv = (const float*)d_in[8];
  const float* Wo = (const float*)d_in[9];
  const float* bo = (const float*)d_in[10];
  float* out = (float*)d_out;

  bf16_t* q_ws = (bf16_t*)d_ws;                        // [B,H,S,64]
  bf16_t* k_ws = q_ws + (size_t)MROWS * DM;            // [B,H,S,64]
  bf16_t* v_ws = k_ws + (size_t)MROWS * DM;            // [B,H,64,S]
  bf16_t* o_ws = v_ws + (size_t)MROWS * DM;            // [B,S,1024]

  dim3 ggrid(DM / 128, MROWS / 128);
  dim3 gblk(256);
  const float qscale = 0.125f;                         // 1/sqrt(64)

  gemm_proj<false><<<ggrid, gblk, 0, stream>>>(query, nullptr, Wq, bq, q_ws, nullptr, MODE_QK, qscale);
  gemm_proj<false><<<ggrid, gblk, 0, stream>>>(key_,  nullptr, Wk, bk, k_ws, nullptr, MODE_QK, 1.0f);
  gemm_proj<false><<<ggrid, gblk, 0, stream>>>(value, nullptr, Wv, bv, v_ws, nullptr, MODE_V, 1.0f);

  flash_attn<<<dim3(SEQ / 64, BATCH * HEADS), dim3(128), 0, stream>>>(q_ws, k_ws, v_ws, o_ws);

  gemm_proj<true><<<ggrid, gblk, 0, stream>>>(nullptr, o_ws, Wo, bo, nullptr, out, MODE_OUT, 1.0f);
}
